// LoopyLayer_31980326486099
// MI455X (gfx1250) — matrix operand order repaired and checked by reading, not compiled, not run on hardware
//
#include <hip/hip_runtime.h>

typedef _Float16 h8   __attribute__((ext_vector_type(8)));
typedef _Float16 v16h __attribute__((ext_vector_type(16)));
typedef float    v8f  __attribute__((ext_vector_type(8)));

#define C 128

// D(16x16,f32) += A(16x128 f16 LDS, row-major) @ B(128x128 f16 global, col-major BT[n*C+k])
// Fragment layouts per CDNA5 ISA 7.12.2:
//   A 16x32 f16 : lane l -> M=l&15, halves 0..7 -> K=koff+j, halves 8..15 -> K=16+koff+j (koff=8*(l>>4))
//   B 32x16 f16 : lane l -> N=l&15, halves 0..7 -> K=koff+j, halves 8..15 -> K=16+koff+j (koff=8*(l>>4))
//   C/D 16x16 f32: lane l -> N=l&15, VGPR v -> M = v + 8*(l>>4)
__device__ inline v8f wmma_gemm_16x16(const _Float16* __restrict__ sA,
                                      const _Float16* __restrict__ BT,
                                      int nb, int lane, v8f acc) {
  const int m    = lane & 15;
  const int koff = (lane >> 4) * 8;
  const int bn   = nb + (lane & 15);
  const int bko  = (lane >> 4) * 8;
  const _Float16* bcol = BT + bn * C;
#pragma unroll
  for (int kc = 0; kc < 4; ++kc) {
    const int K = kc * 32;
    h8 alo = *(const h8*)(sA + m * C + K + koff);
    h8 ahi = *(const h8*)(sA + m * C + K + 16 + koff);
    v16h a = __builtin_shufflevector(alo, ahi,
                                     0,1,2,3,4,5,6,7,8,9,10,11,12,13,14,15);
    h8 blo = *(const h8*)(bcol + K + bko);
    h8 bhi = *(const h8*)(bcol + K + 16 + bko);
    v16h b = __builtin_shufflevector(blo, bhi,
                                     0,1,2,3,4,5,6,7,8,9,10,11,12,13,14,15);
    acc = __builtin_amdgcn_wmma_f32_16x16x32_f16(false, a, false, b,
                                                 (short)0, acc, false, false);
  }
  return acc;
}

__global__ void k_init(const float* __restrict__ x, _Float16* __restrict__ xh,
                       float* __restrict__ rc, int total) {
  int i = blockIdx.x * blockDim.x + threadIdx.x;
  if (i < total) { float v = x[i]; xh[i] = (_Float16)v; rc[i] = 0.f; }
}

// dst_f16[n*C+k] = src_f32[k*C+n], 128x128 (launch <<<128,128>>>)
__global__ void k_transpose(const float* __restrict__ src, _Float16* __restrict__ dst) {
  int i = blockIdx.x * blockDim.x + threadIdx.x;
  int n = i >> 7, k = i & (C - 1);
  dst[n * C + k] = (_Float16)src[k * C + n];
}

// embW[(r*3+a)*C + n] = sum_k emb[r][a][k] * Wt[r][C+k][n]   (launch <<<9,128>>>)
__global__ void k_embW(const float* __restrict__ emb, const float* __restrict__ Wt,
                       float* __restrict__ embW) {
  int i = blockIdx.x;            // r*3 + a
  int r = i / 3;
  int n = threadIdx.x;
  const float* e  = emb + (size_t)i * C;
  const float* wb = Wt + (size_t)r * 2 * C * C + (size_t)C * C;
  float s = 0.f;
  for (int k = 0; k < C; ++k) s += e[k] * wb[(size_t)k * C + n];
  embW[(size_t)i * C + n] = s;
}

// rc[tgt] += (1+r_eps[0]) * x[src]
__global__ void k_seg0(const float* __restrict__ x, const int* __restrict__ ln0,
                       const float* __restrict__ r_eps, float* __restrict__ rc, int E0n) {
  int gid = blockIdx.x * blockDim.x + threadIdx.x;
  int e = gid >> 5;
  int c0 = (gid & 31) * 4;
  if (e >= E0n) return;
  float s = 1.f + r_eps[0];
  int tgt = ln0[e];
  int src = ln0[E0n + e];
  float4 xv = *(const float4*)(x + (size_t)src * C + c0);
  float* o = rc + (size_t)tgt * C + c0;
  atomicAdd(o + 0, s * xv.x);
  atomicAdd(o + 1, s * xv.y);
  atomicAdd(o + 2, s * xv.z);
  atomicAdd(o + 3, s * xv.w);
}

template<int L>
__global__ __launch_bounds__(128)
void k_round(const _Float16* __restrict__ xh,
             const int* __restrict__ nl,      // (L+2, EL)
             const int* __restrict__ lA,      // (L+2, EL)
             const _Float16* __restrict__ WtTopT,
             const float* __restrict__ bt_i,
             const _Float16* __restrict__ w0T,
             const float* __restrict__ b0_i,
             const _Float16* __restrict__ w1T,
             const float* __restrict__ b1_i,
             const float* __restrict__ embW_i,   // [3][C]
             const float* __restrict__ conv_eps,
             const float* __restrict__ r_eps,
             float* __restrict__ rc,
             int EL_) {
  constexpr int S = L + 1;
  __shared__ __align__(16) _Float16 sCC[S][16 * C];
  __shared__ __align__(16) _Float16 sH [S][16 * C];
  __shared__ int sAT[S][16];
  __shared__ int sTGT[16];

  const int tid  = threadIdx.x;
  const int lane = tid & 31;
  const int wave = tid >> 5;
  const int e0   = blockIdx.x * 16;

  if (tid < 16) {
    int e = min(e0 + tid, EL_ - 1);
    sTGT[tid] = nl[e];
#pragma unroll
    for (int s = 0; s < S; ++s) sAT[s][tid] = lA[(size_t)(1 + s) * EL_ + e];
  }
  {
    const int r = tid >> 3;
    const int p = tid & 7;
    int e = min(e0 + r, EL_ - 1);
#pragma unroll
    for (int s = 0; s < S; ++s) {
      int node = nl[(size_t)(1 + s) * EL_ + e];
      const uint4* src = (const uint4*)(xh + (size_t)node * C + p * 16);
      uint4* dst = (uint4*)(&sCC[s][r * C + p * 16]);
      dst[0] = src[0];
      dst[1] = src[1];
    }
  }
  __syncthreads();

  const float conv_s = 1.f + conv_eps[L - 1];
  const float r_s    = 1.f + r_eps[L];
  const int nlo = lane & 15;
  const int mhi = (lane >> 4) * 8;

  // h[s] = cc[s] @ WtTop + bt + embW[at[s]]
#pragma unroll
  for (int s = 0; s < S; ++s) {
    v8f d0 = {}, d1 = {};
    d0 = wmma_gemm_16x16(sCC[s], WtTopT, wave * 32,      lane, d0);
    d1 = wmma_gemm_16x16(sCC[s], WtTopT, wave * 32 + 16, lane, d1);
#pragma unroll
    for (int t = 0; t < 2; ++t) {
      const int n = wave * 32 + t * 16 + nlo;
      v8f d = t ? d1 : d0;
      float bias = bt_i[n];
#pragma unroll
      for (int v = 0; v < 8; ++v) {
        int m = mhi + v;
        float hv = d[v] + bias + embW_i[sAT[s][m] * C + n];
        sH[s][m * C + n] = (_Float16)hv;
      }
    }
  }
  __syncthreads();

  // a[s] = conv_s*cc[s] + h[s-1] + h[s+1]   (in place into sCC)
#pragma unroll
  for (int s = 0; s < S; ++s) {
    for (int idx = tid; idx < 16 * C; idx += 128) {
      float y = 0.f;
      if (s > 0)     y += (float)sH[s - 1][idx];
      if (s < S - 1) y += (float)sH[s + 1][idx];
      float a = conv_s * (float)sCC[s][idx] + y;
      sCC[s][idx] = (_Float16)a;
    }
  }
  __syncthreads();

  // o = sum_s mlp2(a[s]); accumulate slot sum directly in D fragments
  v8f acc0 = {}, acc1 = {};
#pragma unroll
  for (int s = 0; s < S; ++s) {
    v8f t0 = {}, t1 = {};
    t0 = wmma_gemm_16x16(sCC[s], w0T, wave * 32,      lane, t0);
    t1 = wmma_gemm_16x16(sCC[s], w0T, wave * 32 + 16, lane, t1);
#pragma unroll
    for (int t = 0; t < 2; ++t) {
      const int n = wave * 32 + t * 16 + nlo;
      v8f d = t ? t1 : t0;
      float bias = b0_i[n];
#pragma unroll
      for (int v = 0; v < 8; ++v) {
        float hv = d[v] + bias;
        hv = hv > 0.f ? hv : 0.f;
        sH[s][(mhi + v) * C + n] = (_Float16)hv;
      }
    }
    __syncthreads();
    acc0 = wmma_gemm_16x16(sH[s], w1T, wave * 32,      lane, acc0);
    acc1 = wmma_gemm_16x16(sH[s], w1T, wave * 32 + 16, lane, acc1);
  }

  // rc[tgt] += r_s * (acc + S*b1)
#pragma unroll
  for (int t = 0; t < 2; ++t) {
    const int n = wave * 32 + t * 16 + nlo;
    v8f d = t ? acc1 : acc0;
    float bias = (float)S * b1_i[n];
#pragma unroll
    for (int v = 0; v < 8; ++v) {
      int m = mhi + v;
      atomicAdd(&rc[(size_t)sTGT[m] * C + n], r_s * (d[v] + bias));
    }
  }
}

__global__ __launch_bounds__(128)
void k_fin(const float* __restrict__ x, const float* __restrict__ rc,
           const _Float16* __restrict__ w0T, const float* __restrict__ b0,
           const _Float16* __restrict__ w1T, const float* __restrict__ b1,
           const float* __restrict__ eps, float* __restrict__ out, int Nn) {
  __shared__ __align__(16) _Float16 sA[16 * C];
  __shared__ __align__(16) _Float16 sT[16 * C];
  const int tid = threadIdx.x, lane = tid & 31, wave = tid >> 5;
  const int r0 = blockIdx.x * 16;
  const float es = 1.f + eps[0];
  {
    const int r = tid >> 3, p = tid & 7;
    int row = min(r0 + r, Nn - 1);
    const float4* xv = (const float4*)(x + (size_t)row * C + p * 16);
    const float4* rv = (const float4*)(rc + (size_t)row * C + p * 16);
#pragma unroll
    for (int q = 0; q < 4; ++q) {
      float4 a = xv[q], b = rv[q];
      _Float16* d = &sA[r * C + p * 16 + q * 4];
      d[0] = (_Float16)(es * a.x + b.x);
      d[1] = (_Float16)(es * a.y + b.y);
      d[2] = (_Float16)(es * a.z + b.z);
      d[3] = (_Float16)(es * a.w + b.w);
    }
  }
  __syncthreads();
  const int nlo = lane & 15, mhi = (lane >> 4) * 8;
  v8f t0 = {}, t1 = {};
  t0 = wmma_gemm_16x16(sA, w0T, wave * 32,      lane, t0);
  t1 = wmma_gemm_16x16(sA, w0T, wave * 32 + 16, lane, t1);
#pragma unroll
  for (int t = 0; t < 2; ++t) {
    const int n = wave * 32 + t * 16 + nlo;
    v8f d = t ? t1 : t0;
    float bias = b0[n];
#pragma unroll
    for (int v = 0; v < 8; ++v) {
      float hv = d[v] + bias;
      hv = hv > 0.f ? hv : 0.f;
      sT[(mhi + v) * C + n] = (_Float16)hv;
    }
  }
  __syncthreads();
  v8f o0 = {}, o1 = {};
  o0 = wmma_gemm_16x16(sT, w1T, wave * 32,      lane, o0);
  o1 = wmma_gemm_16x16(sT, w1T, wave * 32 + 16, lane, o1);
#pragma unroll
  for (int t = 0; t < 2; ++t) {
    const int n = wave * 32 + t * 16 + nlo;
    v8f d = t ? o1 : o0;
    float bias = b1[n];
#pragma unroll
    for (int v = 0; v < 8; ++v) {
      int row = r0 + mhi + v;
      if (row < Nn) out[(size_t)row * C + n] = d[v] + bias;
    }
  }
}

extern "C" void kernel_launch(void* const* d_in, const int* in_sizes, int n_in,
                              void* d_out, int out_size, void* d_ws, size_t ws_size,
                              hipStream_t stream) {
  (void)n_in; (void)out_size; (void)ws_size;
  const float* x   = (const float*)d_in[0];
  const int* ln0   = (const int*)d_in[1];
  const int* ln1   = (const int*)d_in[2];
  const int* la1   = (const int*)d_in[3];
  const int* ln2   = (const int*)d_in[4];
  const int* la2   = (const int*)d_in[5];
  const int* ln3   = (const int*)d_in[6];
  const int* la3   = (const int*)d_in[7];
  const float* emb = (const float*)d_in[8];
  const float* Wt  = (const float*)d_in[9];
  const float* bt  = (const float*)d_in[10];
  const float* w0  = (const float*)d_in[11];
  const float* b0  = (const float*)d_in[12];
  const float* w1  = (const float*)d_in[13];
  const float* b1  = (const float*)d_in[14];
  const float* conv_eps = (const float*)d_in[15];
  const float* r_eps    = (const float*)d_in[16];
  const float* eps      = (const float*)d_in[17];
  const float* fw0 = (const float*)d_in[18];
  const float* fb0 = (const float*)d_in[19];
  const float* fw1 = (const float*)d_in[20];
  const float* fb1 = (const float*)d_in[21];
  float* out = (float*)d_out;

  const int Nn  = in_sizes[0] / C;   // 100000
  const int E0n = in_sizes[1] / 2;   // 300000
  const int ELn = in_sizes[2] / 3;   // 80000

  char* p = (char*)d_ws;
  _Float16* xh     = (_Float16*)p; p += (size_t)Nn * C * sizeof(_Float16);
  float*    rc     = (float*)p;    p += (size_t)Nn * C * sizeof(float);
  _Float16* WtTopT = (_Float16*)p; p += (size_t)3 * C * C * sizeof(_Float16);
  _Float16* w0T    = (_Float16*)p; p += (size_t)3 * C * C * sizeof(_Float16);
  _Float16* w1T    = (_Float16*)p; p += (size_t)3 * C * C * sizeof(_Float16);
  _Float16* fw0T   = (_Float16*)p; p += (size_t)C * C * sizeof(_Float16);
  _Float16* fw1T   = (_Float16*)p; p += (size_t)C * C * sizeof(_Float16);
  float*    embW   = (float*)p;    p += (size_t)9 * C * sizeof(float);

  const int total = Nn * C;
  k_init<<<(total + 255) / 256, 256, 0, stream>>>(x, xh, rc, total);

  for (int i = 0; i < 3; ++i) {
    k_transpose<<<C, C, 0, stream>>>(Wt + (size_t)i * 2 * C * C, WtTopT + (size_t)i * C * C);
    k_transpose<<<C, C, 0, stream>>>(w0 + (size_t)i * C * C,     w0T    + (size_t)i * C * C);
    k_transpose<<<C, C, 0, stream>>>(w1 + (size_t)i * C * C,     w1T    + (size_t)i * C * C);
  }
  k_transpose<<<C, C, 0, stream>>>(fw0, fw0T);
  k_transpose<<<C, C, 0, stream>>>(fw1, fw1T);
  k_embW<<<9, C, 0, stream>>>(emb, Wt, embW);

  k_seg0<<<(E0n * 32 + 255) / 256, 256, 0, stream>>>(x, ln0, r_eps, rc, E0n);

  const int tiles = (ELn + 15) / 16;
  k_round<1><<<tiles, 128, 0, stream>>>(xh, ln1, la1,
      WtTopT,             bt,        w0T,             b0,        w1T,             b1,
      embW,               conv_eps, r_eps, rc, ELn);
  k_round<2><<<tiles, 128, 0, stream>>>(xh, ln2, la2,
      WtTopT + C * C,     bt + C,    w0T + C * C,     b0 + C,    w1T + C * C,     b1 + C,
      embW + 3 * C,       conv_eps, r_eps, rc, ELn);
  k_round<3><<<tiles, 128, 0, stream>>>(xh, ln3, la3,
      WtTopT + 2 * C * C, bt + 2*C,  w0T + 2 * C * C, b0 + 2*C,  w1T + 2 * C * C, b1 + 2*C,
      embW + 6 * C,       conv_eps, r_eps, rc, ELn);

  k_fin<<<(Nn + 15) / 16, 128, 0, stream>>>(x, rc, fw0T, fb0, fw1T, fb1, eps, out, Nn);
}
